// MultiHeadedAttention_84104049590515
// MI455X (gfx1250) — matrix operand order repaired and checked by reading, compiled, not run on hardware
//
#include <hip/hip_runtime.h>
#include <hip/hip_bf16.h>

// ---------------- problem constants ----------------
#define BATCH   2
#define S_LEN   2048
#define DMODEL  1024
#define NHEAD   16
#define DK      64
#define NEGINF  (-1000000000.0f)

typedef __attribute__((ext_vector_type(16))) __bf16 bf16x16;
typedef __attribute__((ext_vector_type(8)))  __bf16 bf16x8;
typedef __attribute__((ext_vector_type(8)))  float  v8f;

// D = A*B + C, bf16 inputs, f32 accumulate
__device__ inline v8f wmma_bf16(bf16x16 a, bf16x16 b, v8f c) {
  return __builtin_amdgcn_wmma_f32_16x16x32_bf16(
      /*neg_a=*/false, a, /*neg_b=*/false, b,
      /*c_mod=*/(short)0, c, /*reuse_a=*/false, /*reuse_b=*/false);
}

// A-matrix 16x32 bf16 fragment from a row-major LDS row:
// element e in 0..7  -> k = 8*half + e ; e in 8..15 -> k = 16 + 8*half + (e-8)
__device__ inline bf16x16 fragA_row(const __bf16* row, int kh) {
  bf16x8 lo = *(const bf16x8*)(row + kh * 8);
  bf16x8 hi = *(const bf16x8*)(row + 16 + kh * 8);
  bf16x16 r;
#pragma unroll
  for (int i = 0; i < 8; ++i) { r[i] = lo[i]; r[8 + i] = hi[i]; }
  return r;
}

// B-matrix 32x16 bf16 fragment; per lane a column, e in 0..7 -> k = 8*half + e ; e in 8..15 -> k = 16 + 8*half + (e-8)
__device__ inline bf16x16 fragB_row(const __bf16* row, int kh) {
  bf16x8 lo = *(const bf16x8*)(row + kh * 8);
  bf16x8 hi = *(const bf16x8*)(row + 16 + kh * 8);
  bf16x16 r;
#pragma unroll
  for (int i = 0; i < 8; ++i) { r[i] = lo[i]; r[8 + i] = hi[i]; }
  return r;
}

// wave-local LDS producer->consumer sync (lanes of the same wave)
__device__ inline void wave_lds_sync() {
  asm volatile("s_wait_dscnt 0" ::: "memory");
  __builtin_amdgcn_wave_barrier();
}

// ---- CDNA5 async global->LDS copy (ASYNCcnt-tracked DMA), 16B per lane ----
// LDS destination address = low 32 bits of the generic pointer (LDS aperture
// truncation, ISA 10.2). One instruction per wave; every lane copies 16B.
__device__ inline void async_g2l_b128(void* lds, const void* gaddr) {
  unsigned ldsOff = (unsigned)(unsigned long long)lds;
  unsigned long long ga = (unsigned long long)gaddr;
  asm volatile("global_load_async_to_lds_b128 %0, %1, off"
               :: "v"(ldsOff), "v"(ga) : "memory");
}
__device__ inline void wait_async0() {
  asm volatile("s_wait_asynccnt 0" ::: "memory");
}

// ============================================================
// Kernel 1: Y = X @ W^T + b ; X fp32 [B*S, D], W fp32 [D, D] (row n, col k)
// output bf16 in [B, H, S, dk] layout.
// 256 threads (8 waves); macro tile 128x64; wave tile 16x64.
// Double-buffered LDS; next tile's global loads are in flight during WMMAs.
// ============================================================
__global__ __launch_bounds__(256) void proj_gemm_kernel(
    const float* __restrict__ X, const float* __restrict__ W,
    const float* __restrict__ bias, __bf16* __restrict__ out) {
  __shared__ alignas(16) __bf16 As[2][128][32];
  __shared__ alignas(16) __bf16 Ws[2][64][32];

  const int tid  = threadIdx.x;
  const int lane = tid & 31, wave = tid >> 5;
  const int l15  = lane & 15, kh = lane >> 4;
  const int rowBase = blockIdx.x * 128;  // rows of X (B*S flattened)
  const int colBase = blockIdx.y * 64;   // output feature n

  const int ar = tid >> 1, ac = (tid & 1) * 16;
  const int wr = tid >> 2, wc = (tid & 3) * 8;
  const float* Arow = X + (size_t)(rowBase + ar) * DMODEL + ac;
  const float* Wrow = W + (size_t)(colBase + wr) * DMODEL + wc;

  // prologue: stage tile 0 into registers
  float4 xa0, xa1, xa2, xa3, xw0, xw1;
  {
    const float4* xs = (const float4*)(Arow);
    xa0 = xs[0]; xa1 = xs[1]; xa2 = xs[2]; xa3 = xs[3];
    const float4* ws = (const float4*)(Wrow);
    xw0 = ws[0]; xw1 = ws[1];
  }

  v8f acc0 = {}, acc1 = {}, acc2 = {}, acc3 = {};
  const int NIT = DMODEL / 32;

  for (int it = 0; it < NIT; ++it) {
    const int buf = it & 1;
    // convert staged registers -> LDS (bf16)
    {
      __bf16* da = &As[buf][ar][ac];
      da[ 0]=(__bf16)xa0.x; da[ 1]=(__bf16)xa0.y; da[ 2]=(__bf16)xa0.z; da[ 3]=(__bf16)xa0.w;
      da[ 4]=(__bf16)xa1.x; da[ 5]=(__bf16)xa1.y; da[ 6]=(__bf16)xa1.z; da[ 7]=(__bf16)xa1.w;
      da[ 8]=(__bf16)xa2.x; da[ 9]=(__bf16)xa2.y; da[10]=(__bf16)xa2.z; da[11]=(__bf16)xa2.w;
      da[12]=(__bf16)xa3.x; da[13]=(__bf16)xa3.y; da[14]=(__bf16)xa3.z; da[15]=(__bf16)xa3.w;
      __bf16* dw = &Ws[buf][wr][wc];
      dw[0]=(__bf16)xw0.x; dw[1]=(__bf16)xw0.y; dw[2]=(__bf16)xw0.z; dw[3]=(__bf16)xw0.w;
      dw[4]=(__bf16)xw1.x; dw[5]=(__bf16)xw1.y; dw[6]=(__bf16)xw1.z; dw[7]=(__bf16)xw1.w;
    }
    __syncthreads();
    // issue next tile's global loads (in flight during WMMAs)
    if (it + 1 < NIT) {
      const float4* xs = (const float4*)(Arow + (it + 1) * 32);
      xa0 = xs[0]; xa1 = xs[1]; xa2 = xs[2]; xa3 = xs[3];
      const float4* ws = (const float4*)(Wrow + (it + 1) * 32);
      xw0 = ws[0]; xw1 = ws[1];
    }
    bf16x16 a  = fragA_row(&As[buf][wave * 16 + l15][0], kh);
    bf16x16 b0 = fragB_row(&Ws[buf][ 0 + l15][0], kh);
    bf16x16 b1 = fragB_row(&Ws[buf][16 + l15][0], kh);
    bf16x16 b2 = fragB_row(&Ws[buf][32 + l15][0], kh);
    bf16x16 b3 = fragB_row(&Ws[buf][48 + l15][0], kh);
    acc0 = wmma_bf16(a, b0, acc0);
    acc1 = wmma_bf16(a, b1, acc1);
    acc2 = wmma_bf16(a, b2, acc2);
    acc3 = wmma_bf16(a, b3, acc3);
  }

  v8f* accs[4] = {&acc0, &acc1, &acc2, &acc3};
#pragma unroll
  for (int ct = 0; ct < 4; ++ct) {
    const int nG = colBase + ct * 16 + l15;
    const float bn = bias[nG];
    const int h = nG >> 6, dd = nG & 63;
#pragma unroll
    for (int r = 0; r < 8; ++r) {
      const int rowG = rowBase + wave * 16 + kh * 8 + r;
      const int bb = rowG / S_LEN, ss = rowG % S_LEN;
      const size_t idx = ((((size_t)bb * NHEAD + h) * S_LEN) + ss) * DK + dd;
      out[idx] = (__bf16)((*accs[ct])[r] + bn);
    }
  }
}

// ============================================================
// Kernel 2: flash attention (S^T = K·Q^T trick). K tiles double-buffered via
// CDNA5 async global->LDS DMA; V tiles double-buffered through registers.
// One barrier per 32-key step.
// ============================================================
__global__ __launch_bounds__(128) void attn_kernel(
    const __bf16* __restrict__ Q, const __bf16* __restrict__ K,
    const __bf16* __restrict__ V, const int* __restrict__ mask,
    __bf16* __restrict__ ctx) {
  __shared__ alignas(16) __bf16 Ks[2][32][DK];    // [key][dk]
  __shared__ alignas(16) __bf16 Vt[2][DK][32];    // [dk][key] (transposed)
  __shared__ alignas(16) __bf16 Qs[4][16][DK];    // per-wave [q][dk]
  __shared__ alignas(16) float  alpha_s[4][16];
  __shared__ alignas(16) float  lden_s[4][16];

  const int tid  = threadIdx.x;
  const int lane = tid & 31, wave = tid >> 5;
  const int l15  = lane & 15, kh = lane >> 4;
  const int by = blockIdx.y;
  const int b = by >> 4, h = by & 15;
  const int qbase = blockIdx.x * 64 + wave * 16;

  const __bf16* qptr  = Q + (((size_t)b * NHEAD + h) * S_LEN + qbase) * DK;
  const __bf16* kbase = K + (((size_t)b * NHEAD + h) * S_LEN) * DK;
  const __bf16* vbase = V + (((size_t)b * NHEAD + h) * S_LEN) * DK;
  const int* kmaskb = mask + b * S_LEN;

  // stage this wave's Q tile (16x64 bf16) via async DMA (waited at iter 0)
  {
    const int r = lane >> 1, off = (lane & 1) * 32;
    const __bf16* src = qptr + r * DK + off;
    async_g2l_b128(&Qs[wave][r][off],      src);
    async_g2l_b128(&Qs[wave][r][off + 8],  src + 8);
    async_g2l_b128(&Qs[wave][r][off + 16], src + 16);
    async_g2l_b128(&Qs[wave][r][off + 24], src + 24);
  }

  const int sr  = tid >> 2;            // staged key row 0..31
  const int soff = (tid & 3) * 16;     // staged dk offset

  // prologue: async K tile 0, V tile 0 into registers
  {
    const __bf16* kg = kbase + (size_t)sr * DK + soff;
    async_g2l_b128(&Ks[0][sr][soff],     kg);
    async_g2l_b128(&Ks[0][sr][soff + 8], kg + 8);
  }
  uint4 vr0 = *(const uint4*)(vbase + (size_t)sr * DK + soff);
  uint4 vr1 = *(const uint4*)(vbase + (size_t)sr * DK + soff + 8);

  const int qmask = kmaskb[qbase + l15];   // mask for this lane's q column

  float m_run = -3.0e38f;
  float l_run = 0.0f;
  v8f acc0 = {}, acc1 = {}, acc2 = {}, acc3 = {};

  const int NIT = S_LEN / 32;
  for (int it = 0; it < NIT; ++it) {
    const int buf = it & 1;
    const int kb = it * 32;
    // transpose-store staged V registers into Vt[buf]
    {
      __bf16 tmp[16];
      *(uint4*)&tmp[0] = vr0;
      *(uint4*)&tmp[8] = vr1;
#pragma unroll
      for (int i = 0; i < 16; ++i) Vt[buf][soff + i][sr] = tmp[i];
    }
    wait_async0();           // Ks[buf] (this wave's DMA) complete
    __syncthreads();
    // issue next tile's DMA + loads (in flight during WMMAs/softmax)
    if (it + 1 < NIT) {
      const __bf16* kg = kbase + (size_t)(kb + 32 + sr) * DK + soff;
      async_g2l_b128(&Ks[buf ^ 1][sr][soff],     kg);
      async_g2l_b128(&Ks[buf ^ 1][sr][soff + 8], kg + 8);
      vr0 = *(const uint4*)(vbase + (size_t)(kb + 32 + sr) * DK + soff);
      vr1 = *(const uint4*)(vbase + (size_t)(kb + 32 + sr) * DK + soff + 8);
      if (kb + 64 < S_LEN)
        __builtin_prefetch(vbase + (size_t)(kb + 64) * DK, 0, 1);
    }

    // ---- S^T tiles: rows = keys, cols = q ----
    v8f st0 = {}, st1 = {};
    {
      bf16x16 bQ0 = fragB_row(&Qs[wave][l15][0],  kh);
      bf16x16 bQ1 = fragB_row(&Qs[wave][l15][32], kh);
      bf16x16 aK0 = fragA_row(&Ks[buf][l15][0],   kh);
      bf16x16 aK1 = fragA_row(&Ks[buf][l15][32],  kh);
      st0 = wmma_bf16(aK0, bQ0, st0);
      st0 = wmma_bf16(aK1, bQ1, st0);
      bf16x16 aK2 = fragA_row(&Ks[buf][16 + l15][0],  kh);
      bf16x16 aK3 = fragA_row(&Ks[buf][16 + l15][32], kh);
      st1 = wmma_bf16(aK2, bQ0, st1);
      st1 = wmma_bf16(aK3, bQ1, st1);
    }

    // ---- mask + scale, online softmax ----
    const float sc = 0.125f;  // 1/sqrt(64)
    float p0[8], p1[8];
    float tmax = -3.0e38f;
#pragma unroll
    for (int r = 0; r < 8; ++r) {
      const int km0 = kmaskb[kb + kh * 8 + r];
      const int km1 = kmaskb[kb + 16 + kh * 8 + r];
      const float s0 = (km0 && qmask) ? st0[r] * sc : NEGINF;
      const float s1 = (km1 && qmask) ? st1[r] * sc : NEGINF;
      p0[r] = s0; p1[r] = s1;
      tmax = fmaxf(tmax, fmaxf(s0, s1));
    }
    tmax = fmaxf(tmax, __shfl_xor(tmax, 16, 32));
    const float m_new = fmaxf(m_run, tmax);
    const float alpha = __expf(m_run - m_new);
    m_run = m_new;

    bf16x16 aP;
    float lsum = 0.0f;
#pragma unroll
    for (int r = 0; r < 8; ++r) {
      const float e0 = __expf(p0[r] - m_new);
      const float e1 = __expf(p1[r] - m_new);
      lsum += e0 + e1;
      aP[r]     = (__bf16)e0;   // keys kb + 8*kh + r      (A elems 0..7)
      aP[8 + r] = (__bf16)e1;   // keys kb + 16 + 8*kh + r (A elems 8..15)
    }
    lsum += __shfl_xor(lsum, 16, 32);
    l_run = l_run * alpha + lsum;

    // broadcast per-q alpha to per-row consumers via LDS (wave-local)
    alpha_s[wave][l15] = alpha;
    wave_lds_sync();
    float ar8[8];
#pragma unroll
    for (int r = 0; r < 8; ++r) ar8[r] = alpha_s[wave][kh * 8 + r];
#pragma unroll
    for (int r = 0; r < 8; ++r) {
      acc0[r] *= ar8[r]; acc1[r] *= ar8[r];
      acc2[r] *= ar8[r]; acc3[r] *= ar8[r];
    }

    // ---- ctx += P @ V (4 dim tiles of 16) ----
    bf16x16 bV0 = fragB_row(&Vt[buf][ 0 + l15][0], kh);
    bf16x16 bV1 = fragB_row(&Vt[buf][16 + l15][0], kh);
    bf16x16 bV2 = fragB_row(&Vt[buf][32 + l15][0], kh);
    bf16x16 bV3 = fragB_row(&Vt[buf][48 + l15][0], kh);
    acc0 = wmma_bf16(aP, bV0, acc0);
    acc1 = wmma_bf16(aP, bV1, acc1);
    acc2 = wmma_bf16(aP, bV2, acc2);
    acc3 = wmma_bf16(aP, bV3, acc3);
  }

  // final 1/l per row, then write ctx [B, S, H*dk]
  lden_s[wave][l15] = l_run;
  wave_lds_sync();
  float linv[8];
#pragma unroll
  for (int r = 0; r < 8; ++r) linv[r] = 1.0f / lden_s[wave][kh * 8 + r];

  const size_t rowb = (size_t)b * S_LEN + qbase;
#pragma unroll
  for (int r = 0; r < 8; ++r) {
    __bf16* o = ctx + (rowb + kh * 8 + r) * DMODEL + h * DK;
    o[ 0 + l15] = (__bf16)(acc0[r] * linv[r]);
    o[16 + l15] = (__bf16)(acc1[r] * linv[r]);
    o[32 + l15] = (__bf16)(acc2[r] * linv[r]);
    o[48 + l15] = (__bf16)(acc3[r] * linv[r]);
  }
}

// ============================================================
// Kernel 3: out = ctx @ wo^T + bo ; ctx bf16 [B*S, D], out fp32 [B*S, D].
// A tile double-buffered via async DMA; W staged through registers.
// ============================================================
__global__ __launch_bounds__(256) void out_gemm_kernel(
    const __bf16* __restrict__ Xc, const float* __restrict__ W,
    const float* __restrict__ bias, float* __restrict__ out) {
  __shared__ alignas(16) __bf16 As[2][128][32];
  __shared__ alignas(16) __bf16 Ws[2][64][32];

  const int tid  = threadIdx.x;
  const int lane = tid & 31, wave = tid >> 5;
  const int l15  = lane & 15, kh = lane >> 4;
  const int rowBase = blockIdx.x * 128;
  const int colBase = blockIdx.y * 64;

  const int ar = tid >> 1, ac = (tid & 1) * 16;
  const int wr = tid >> 2, wc = (tid & 3) * 8;
  const __bf16* Arow = Xc + (size_t)(rowBase + ar) * DMODEL + ac;
  const float*  Wrow = W  + (size_t)(colBase + wr) * DMODEL + wc;

  // prologue: async A tile 0, W tile 0 into registers
  async_g2l_b128(&As[0][ar][ac],     Arow);
  async_g2l_b128(&As[0][ar][ac + 8], Arow + 8);
  float4 xw0, xw1;
  { const float4* ws = (const float4*)(Wrow); xw0 = ws[0]; xw1 = ws[1]; }

  v8f acc0 = {}, acc1 = {}, acc2 = {}, acc3 = {};
  const int NIT = DMODEL / 32;

  for (int it = 0; it < NIT; ++it) {
    const int buf = it & 1;
    {
      __bf16* dw = &Ws[buf][wr][wc];
      dw[0]=(__bf16)xw0.x; dw[1]=(__bf16)xw0.y; dw[2]=(__bf16)xw0.z; dw[3]=(__bf16)xw0.w;
      dw[4]=(__bf16)xw1.x; dw[5]=(__bf16)xw1.y; dw[6]=(__bf16)xw1.z; dw[7]=(__bf16)xw1.w;
    }
    wait_async0();           // As[buf] (this wave's DMA) complete
    __syncthreads();
    if (it + 1 < NIT) {
      async_g2l_b128(&As[buf ^ 1][ar][ac],     Arow + (it + 1) * 32);
      async_g2l_b128(&As[buf ^ 1][ar][ac + 8], Arow + (it + 1) * 32 + 8);
      const float4* ws = (const float4*)(Wrow + (it + 1) * 32);
      xw0 = ws[0]; xw1 = ws[1];
    }
    bf16x16 a  = fragA_row(&As[buf][wave * 16 + l15][0], kh);
    bf16x16 b0 = fragB_row(&Ws[buf][ 0 + l15][0], kh);
    bf16x16 b1 = fragB_row(&Ws[buf][16 + l15][0], kh);
    bf16x16 b2 = fragB_row(&Ws[buf][32 + l15][0], kh);
    bf16x16 b3 = fragB_row(&Ws[buf][48 + l15][0], kh);
    acc0 = wmma_bf16(a, b0, acc0);
    acc1 = wmma_bf16(a, b1, acc1);
    acc2 = wmma_bf16(a, b2, acc2);
    acc3 = wmma_bf16(a, b3, acc3);
  }

  v8f* accs[4] = {&acc0, &acc1, &acc2, &acc3};
#pragma unroll
  for (int ct = 0; ct < 4; ++ct) {
    const int nG = colBase + ct * 16 + l15;
    const float bn = bias[nG];
#pragma unroll
    for (int r = 0; r < 8; ++r) {
      const int rowG = rowBase + wave * 16 + kh * 8 + r;
      out[(size_t)rowG * DMODEL + nG] = (*accs[ct])[r] + bn;
    }
  }
}

// ============================================================
extern "C" void kernel_launch(void* const* d_in, const int* in_sizes, int n_in,
                              void* d_out, int out_size, void* d_ws, size_t ws_size,
                              hipStream_t stream) {
  const float* query = (const float*)d_in[0];
  const float* key   = (const float*)d_in[1];
  const float* value = (const float*)d_in[2];
  const int*   mask  = (const int*)  d_in[3];
  const float* wq = (const float*)d_in[4];
  const float* bq = (const float*)d_in[5];
  const float* wk = (const float*)d_in[6];
  const float* bk = (const float*)d_in[7];
  const float* wv = (const float*)d_in[8];
  const float* bv = (const float*)d_in[9];
  const float* wo = (const float*)d_in[10];
  const float* bo = (const float*)d_in[11];
  float* out = (float*)d_out;

  const size_t BSD = (size_t)BATCH * S_LEN * DMODEL;  // 4M elems
  __bf16* qws = (__bf16*)d_ws;
  __bf16* kws = qws + BSD;
  __bf16* vws = kws + BSD;
  __bf16* cws = vws + BSD;

  dim3 gGemm(BATCH * S_LEN / 128, DMODEL / 64);  // (32, 16)
  dim3 blkG(256);
  hipLaunchKernelGGL(proj_gemm_kernel, gGemm, blkG, 0, stream, query, wq, bq, qws);
  hipLaunchKernelGGL(proj_gemm_kernel, gGemm, blkG, 0, stream, key,   wk, bk, kws);
  hipLaunchKernelGGL(proj_gemm_kernel, gGemm, blkG, 0, stream, value, wv, bv, vws);

  dim3 gAttn(S_LEN / 64, BATCH * NHEAD);         // (32, 32)
  dim3 blkA(128);
  hipLaunchKernelGGL(attn_kernel, gAttn, blkA, 0, stream, qws, kws, vws, mask, cws);

  hipLaunchKernelGGL(out_gemm_kernel, gGemm, blkG, 0, stream, cws, wo, bo, out);
}
